// SMOKEPredictor_41369124995689
// MI455X (gfx1250) — compile-verified, matrix operand order checked by reading, not run on hardware
//
#include <hip/hip_runtime.h>
#include <hip/hip_bf16.h>

typedef _Float16 v16h __attribute__((ext_vector_type(16)));
typedef float    v8f  __attribute__((ext_vector_type(8)));

// K-local index held by (lane, half-slot) for the 16-bit 16x32 WMMA A layout
// (ISA 7.12.2): lanes 0-15 hold K 0..7 (v0-3) and 16..23 (v4-7); lanes 16-31
// hold K 8..15 and 24..31. B (32x16) assumed symmetric with N<->M.
__device__ __forceinline__ int klocal_of(int lane, int h) {
    int k = (h < 8) ? h : (h + 8);
    if (lane >= 16) k += 8;
    return k;
}

// ---------------------------------------------------------------------------
// Kernel 1: pack cls_w1 (tiles 0..15) and reg_w1 (tiles 16..31) [512 x 576]
// into per-(tile,kstep) WMMA A fragments: Wpack[(t*18+s)*512 + lane*16 + h].
// ---------------------------------------------------------------------------
__global__ void pack_weights(const float* __restrict__ cls_w1,
                             const float* __restrict__ reg_w1,
                             _Float16* __restrict__ Wpack) {
    int idx = blockIdx.x * blockDim.x + threadIdx.x;
    if (idx >= 32 * 18 * 512) return;
    int t    = idx / (18 * 512);
    int r    = idx % (18 * 512);
    int s    = r / 512;
    int q    = r % 512;
    int lane = q / 16, h = q % 16;
    int m    = lane & 15;
    int k    = s * 32 + klocal_of(lane, h);          // 0..575 = ci*9+ky*3+kx
    int oc   = t * 16 + m;
    float w  = (t < 16) ? cls_w1[oc * 576 + k] : reg_w1[(oc - 256) * 576 + k];
    Wpack[idx] = (_Float16)w;
}

// ---------------------------------------------------------------------------
// Kernel 2: fused cls branch: 3x3 conv(64->256) + bias + BN + ReLU + 1x1
// conv(256->3) + sigmoid + clip -> head_class[8,3,96,320].
// Block = 256 thr = 8 waves; block owns (b, y, 32-pixel strip).
// Waves 0-3: pixel group 0, 4 M-tiles each (all 256 cls ch).
// Waves 4-7: pixel group 1, same tiles.
// B fragments pre-swizzled once per block into WMMA lane layout so the inner
// loop reads B with one 32B LDS access instead of 16 ds_load_u16.
// ---------------------------------------------------------------------------
__global__ void __launch_bounds__(256)
conv_cls_fused(const float* __restrict__ in4,
               const _Float16* __restrict__ Wpack,
               const float* __restrict__ cls_b1,
               const float* __restrict__ bn1_g, const float* __restrict__ bn1_b,
               const float* __restrict__ bn1_m, const float* __restrict__ bn1_v,
               const float* __restrict__ cls_w2, const float* __restrict__ cls_b2,
               float* __restrict__ head_class) {
    __shared__ __attribute__((aligned(32))) _Float16 s_in[64 * 102];     // [ci][dy 0..2][xi 0..33]
    __shared__ __attribute__((aligned(32))) _Float16 s_bfrag[2 * 18 * 512];
    __shared__ float s_p[256 * 3];                                        // 1x1 partials

    const int tid  = threadIdx.x;
    const int lane = tid & 31, wave = tid >> 5;
    const int y = blockIdx.y, b = blockIdx.z;
    const int x0 = blockIdx.x * 32;

    // Stage f32 -> f16 input halo tile into LDS (zero-padded).
    for (int i = tid; i < 64 * 102; i += 256) {
        int ci = i / 102, r = i % 102, dy = r / 34, xi = r % 34;
        int yy = y + dy - 1, xx = x0 + xi - 1;
        float v = 0.f;
        if (yy >= 0 && yy < 96 && xx >= 0 && xx < 320)
            v = in4[((b * 64 + ci) * 96 + yy) * 320 + xx];
        s_in[i] = (_Float16)v;
    }
    __syncthreads();

    // Cooperative im2col swizzle into WMMA-B lane layout, once per block.
    // item = (g, ks, lane2): build 16 halves for one lane of one fragment.
    for (int item = tid; item < 2 * 18 * 32; item += 256) {
        int g = item / 576, r = item % 576;
        int ks = r / 32, lane2 = r % 32;
        int n = lane2 & 15;
        v16h tmp;
        #pragma unroll
        for (int h = 0; h < 16; ++h) {
            int k  = ks * 32 + klocal_of(lane2, h);
            int ci = k / 9, rem = k % 9, ky = rem / 3, kx = rem - ky * 3;
            tmp[h] = s_in[ci * 102 + ky * 34 + g * 16 + n + kx];
        }
        *(v16h*)(s_bfrag + (size_t)item * 16) = tmp;
    }
    __syncthreads();

    const int g  = wave >> 2;            // pixel group 0/1
    const int tb = (wave & 3) * 4;       // first of 4 cls tiles
    v8f acc0 = {}, acc1 = {}, acc2 = {}, acc3 = {};

    #pragma unroll 2
    for (int ks = 0; ks < 18; ++ks) {
        v16h bf = *(const v16h*)(s_bfrag + ((g * 18 + ks) * 32 + lane) * 16);
        v16h a0 = *(const v16h*)(Wpack + (((tb + 0) * 18 + ks) * 512 + lane * 16));
        acc0 = __builtin_amdgcn_wmma_f32_16x16x32_f16(false, a0, false, bf,
                                                      (short)0, acc0, false, false);
        v16h a1 = *(const v16h*)(Wpack + (((tb + 1) * 18 + ks) * 512 + lane * 16));
        acc1 = __builtin_amdgcn_wmma_f32_16x16x32_f16(false, a1, false, bf,
                                                      (short)0, acc1, false, false);
        v16h a2 = *(const v16h*)(Wpack + (((tb + 2) * 18 + ks) * 512 + lane * 16));
        acc2 = __builtin_amdgcn_wmma_f32_16x16x32_f16(false, a2, false, bf,
                                                      (short)0, acc2, false, false);
        v16h a3 = *(const v16h*)(Wpack + (((tb + 3) * 18 + ks) * 512 + lane * 16));
        acc3 = __builtin_amdgcn_wmma_f32_16x16x32_f16(false, a3, false, bf,
                                                      (short)0, acc3, false, false);
    }

    // Epilogue: bias + BN + ReLU, then partial 1x1 (256->3) per thread.
    float p0 = 0.f, p1 = 0.f, p2 = 0.f;
    const int rofs = (lane >> 4) * 8;    // C layout: M = r + 8*(lane/16)
    #pragma unroll
    for (int j = 0; j < 4; ++j) {
        v8f a = (j == 0) ? acc0 : (j == 1) ? acc1 : (j == 2) ? acc2 : acc3;
        #pragma unroll
        for (int r = 0; r < 8; ++r) {
            int ch = (tb + j) * 16 + r + rofs;
            float sc = bn1_g[ch] * rsqrtf(bn1_v[ch] + 1e-5f);
            float v  = (a[r] + cls_b1[ch]) * sc + (bn1_b[ch] - bn1_m[ch] * sc);
            v = fmaxf(v, 0.f);
            p0 += cls_w2[ch] * v; p1 += cls_w2[256 + ch] * v; p2 += cls_w2[512 + ch] * v;
        }
    }
    s_p[tid * 3 + 0] = p0; s_p[tid * 3 + 1] = p1; s_p[tid * 3 + 2] = p2;
    __syncthreads();

    // Deterministic reduce: each pixel column is shared by 8 threads
    // (4 waves x 2 half-lanes) within its pixel group.
    if (tid < 96) {
        int o = tid / 32, col = tid % 32;      // col: 0..31 within block strip
        int gg = col / 16, nn = col % 16;
        float s = 0.f;
        for (int t = gg * 128 + nn; t < (gg + 1) * 128; t += 16) s += s_p[t * 3 + o];
        float v  = s + cls_b2[o];
        float sg = 1.f / (1.f + __expf(-v));
        sg = fminf(fmaxf(sg, 1e-4f), 1.f - 1e-4f);
        head_class[((b * 3 + o) * 96 + y) * 320 + x0 + col] = sg;
    }
}

// ---------------------------------------------------------------------------
// Kernel 3: 3x3 NMS (keep where equal to window max, else 0).
// ---------------------------------------------------------------------------
__global__ void nms_kernel(const float* __restrict__ hm, float* __restrict__ out) {
    int i = blockIdx.x * blockDim.x + threadIdx.x;
    if (i >= 8 * 3 * 96 * 320) return;
    int x = i % 320, y = (i / 320) % 96, bc = i / (320 * 96);
    float v = hm[i], mx = -INFINITY;
    for (int dy = -1; dy <= 1; ++dy)
        for (int dx = -1; dx <= 1; ++dx) {
            int yy = y + dy, xx = x + dx;
            if (yy >= 0 && yy < 96 && xx >= 0 && xx < 320)
                mx = fmaxf(mx, hm[(bc * 96 + yy) * 320 + xx]);
        }
    out[i] = (mx == v) ? v : 0.f;
}

// ---------------------------------------------------------------------------
// Kernel 4: per-(b,c) top-50 by iterative argmax (ties -> lowest index).
// Destructive on nmsbuf (rewritten each call by nms_kernel).
// ---------------------------------------------------------------------------
__global__ void __launch_bounds__(256)
topk_per_class(float* __restrict__ nmsbuf, float* __restrict__ tsc, int* __restrict__ tix) {
    __shared__ float sv[256];
    __shared__ int   si[256];
    const int bc = blockIdx.x, tid = threadIdx.x;
    float* base = nmsbuf + (size_t)bc * 30720;
    for (int it = 0; it < 50; ++it) {
        float bv = -INFINITY; int bi = 0x7fffffff;
        for (int i = tid; i < 30720; i += 256) {
            float v = base[i];
            if (v > bv) { bv = v; bi = i; }
        }
        sv[tid] = bv; si[tid] = bi;
        __syncthreads();
        for (int s = 128; s > 0; s >>= 1) {
            if (tid < s) {
                float ov = sv[tid + s]; int oi = si[tid + s];
                if (ov > sv[tid] || (ov == sv[tid] && oi < si[tid])) { sv[tid] = ov; si[tid] = oi; }
            }
            __syncthreads();
        }
        if (tid == 0) {
            tsc[bc * 50 + it] = sv[0];
            tix[bc * 50 + it] = si[0];
            base[si[0]] = -INFINITY;
        }
        __threadfence();
        __syncthreads();
    }
}

// ---------------------------------------------------------------------------
// Kernel 5: combine 3x50 -> global top-50 per batch; emit scores/cls/ys/xs.
// d_out layout: [0,3200) head_reg, [3200) scores, [3600) cls, [4000) ys, [4400) xs.
// ---------------------------------------------------------------------------
__global__ void combine_topk(const float* __restrict__ tsc, const int* __restrict__ tix,
                             float* __restrict__ out) {
    int b = blockIdx.x;
    if (threadIdx.x != 0) return;
    bool used[150];
    for (int i = 0; i < 150; ++i) used[i] = false;
    for (int it = 0; it < 50; ++it) {
        float bv = -INFINITY; int bi = 0;
        for (int i = 0; i < 150; ++i) {
            if (used[i]) continue;
            float v = tsc[b * 150 + i];     // [(b*3+c)*50+j] == b*150 + i
            if (v > bv) { bv = v; bi = i; }
        }
        used[bi] = true;
        int ind = tix[b * 150 + bi];
        out[3200 + b * 50 + it] = bv;
        out[3600 + b * 50 + it] = (float)(bi / 50);
        out[4000 + b * 50 + it] = (float)(ind / 320);
        out[4400 + b * 50 + it] = (float)(ind % 320);
    }
}

// ---------------------------------------------------------------------------
// Kernel 6: build the 4 bilinear corner coords per detection (1600 slots).
// ---------------------------------------------------------------------------
__global__ void build_corners(const float* __restrict__ out, int* __restrict__ sy,
                              int* __restrict__ sx, float* __restrict__ lxv,
                              float* __restrict__ lyv) {
    int i = blockIdx.x * blockDim.x + threadIdx.x;
    if (i >= 400) return;                       // i = b*50 + k
    float x = fminf(fmaxf(out[4400 + i], 0.f), 319.f);
    float y = fminf(fmaxf(out[4000 + i], 0.f), 95.f);
    int x0 = (int)floorf(x), y0 = (int)floorf(y);
    int x1 = min(x0 + 1, 319), y1 = min(y0 + 1, 95);
    lxv[i] = x - (float)x0; lyv[i] = y - (float)y0;
    int base = i * 4;
    sy[base + 0] = y0; sx[base + 0] = x0;
    sy[base + 1] = y0; sx[base + 1] = x1;
    sy[base + 2] = y1; sx[base + 2] = x0;
    sy[base + 3] = y1; sx[base + 3] = x1;
}

// ---------------------------------------------------------------------------
// Kernel 7: reg branch conv(64->256)+BN+ReLU recomputed ONLY at the 1600
// corner pixels (WMMA implicit GEMM; avoids a 252 MB dense intermediate).
// Grid (100, 2); 8 waves/block; wave handles one 16-channel tile x 16 slots.
// ---------------------------------------------------------------------------
__global__ void __launch_bounds__(256)
conv_reg_corners(const float* __restrict__ in4, const _Float16* __restrict__ Wpack,
                 const int* __restrict__ slot_y, const int* __restrict__ slot_x,
                 const float* __restrict__ reg_b1,
                 const float* __restrict__ bn2_g, const float* __restrict__ bn2_b,
                 const float* __restrict__ bn2_m, const float* __restrict__ bn2_v,
                 float* __restrict__ cf) {
    const int tid = threadIdx.x, lane = tid & 31, wave = tid >> 5;
    const int mt   = blockIdx.y * 8 + wave;     // 0..15
    const int tile = 16 + mt;                   // reg tiles in Wpack
    const int nn   = lane & 15;
    const int slot = blockIdx.x * 16 + nn;      // 0..1599
    const int b    = slot / 200;
    const int py = slot_y[slot], px = slot_x[slot];

    v8f acc = {};
    for (int ks = 0; ks < 18; ++ks) {
        v16h bf;
        #pragma unroll
        for (int h = 0; h < 16; ++h) {
            int k  = ks * 32 + klocal_of(lane, h);
            int ci = k / 9, rem = k % 9, ky = rem / 3, kx = rem - ky * 3;
            int yy = py + ky - 1, xx = px + kx - 1;
            float v = 0.f;
            if (yy >= 0 && yy < 96 && xx >= 0 && xx < 320)
                v = in4[((b * 64 + ci) * 96 + yy) * 320 + xx];
            bf[h] = (_Float16)v;
        }
        v16h a = *(const v16h*)(Wpack + ((tile * 18 + ks) * 512 + lane * 16));
        acc = __builtin_amdgcn_wmma_f32_16x16x32_f16(false, a, false, bf,
                                                     (short)0, acc, false, false);
    }
    const int rofs = (lane >> 4) * 8;
    #pragma unroll
    for (int r = 0; r < 8; ++r) {
        int ch = mt * 16 + r + rofs;            // 0..255
        float sc = bn2_g[ch] * rsqrtf(bn2_v[ch] + 1e-5f);
        float v  = (acc[r] + reg_b1[ch]) * sc + (bn2_b[ch] - bn2_m[ch] * sc);
        cf[slot * 256 + ch] = fmaxf(v, 0.f);
    }
}

// ---------------------------------------------------------------------------
// Kernel 8: gather 640-ch ROI vector (p4 from cf, p8/p16 bilinear from
// inputs), 8x640 GEMV, sigmoid/norm postprocess -> head_regression.
// ---------------------------------------------------------------------------
__global__ void __launch_bounds__(256)
final_head(const float* __restrict__ cf, const float* __restrict__ up8,
           const float* __restrict__ up16, const float* __restrict__ lxv,
           const float* __restrict__ lyv, const float* __restrict__ box_w,
           const float* __restrict__ box_b, float* __restrict__ out) {
    __shared__ float s_l[256 * 8];
    __shared__ float reg8[8];
    const int tid = threadIdx.x;
    const int i = blockIdx.x;                   // b*50 + k
    const int b = i / 50, k = i % 50;

    float fx = out[4400 + i], fy = out[4000 + i];
    float lx = lxv[i], ly = lyv[i];
    float w4a = (1.f - ly) * (1.f - lx), w4b = (1.f - ly) * lx;
    float w4c = ly * (1.f - lx),         w4d = ly * lx;

    float x8 = fminf(fmaxf(fx * 0.5f, 0.f), 159.f);
    float y8 = fminf(fmaxf(fy * 0.5f, 0.f), 47.f);
    int x80 = (int)floorf(x8), y80 = (int)floorf(y8);
    int x81 = min(x80 + 1, 159), y81 = min(y80 + 1, 47);
    float l8x = x8 - (float)x80, l8y = y8 - (float)y80;

    float x6 = fminf(fmaxf(fx * 0.25f, 0.f), 79.f);
    float y6 = fminf(fmaxf(fy * 0.25f, 0.f), 23.f);
    int x60 = (int)floorf(x6), y60 = (int)floorf(y6);
    int x61 = min(x60 + 1, 79), y61 = min(y60 + 1, 23);
    float l6x = x6 - (float)x60, l6y = y6 - (float)y60;

    float acc[8];
    #pragma unroll
    for (int o = 0; o < 8; ++o) acc[o] = 0.f;

    for (int c = tid; c < 640; c += 256) {
        float v;
        if (c < 256) {
            int base = i * 4;
            v = w4a * cf[(base + 0) * 256 + c] + w4b * cf[(base + 1) * 256 + c]
              + w4c * cf[(base + 2) * 256 + c] + w4d * cf[(base + 3) * 256 + c];
        } else if (c < 384) {
            const float* f = up8 + (size_t)(b * 128 + (c - 256)) * 48 * 160;
            v = (1.f - l8y) * (1.f - l8x) * f[y80 * 160 + x80]
              + (1.f - l8y) * l8x         * f[y80 * 160 + x81]
              + l8y * (1.f - l8x)         * f[y81 * 160 + x80]
              + l8y * l8x                 * f[y81 * 160 + x81];
        } else {
            const float* f = up16 + (size_t)(b * 256 + (c - 384)) * 24 * 80;
            v = (1.f - l6y) * (1.f - l6x) * f[y60 * 80 + x60]
              + (1.f - l6y) * l6x         * f[y60 * 80 + x61]
              + l6y * (1.f - l6x)         * f[y61 * 80 + x60]
              + l6y * l6x                 * f[y61 * 80 + x61];
        }
        #pragma unroll
        for (int o = 0; o < 8; ++o) acc[o] += box_w[o * 640 + c] * v;
    }
    #pragma unroll
    for (int o = 0; o < 8; ++o) s_l[tid * 8 + o] = acc[o];
    __syncthreads();
    if (tid < 8) {
        float s = 0.f;
        for (int t = 0; t < 256; ++t) s += s_l[t * 8 + tid];
        reg8[tid] = s + box_b[tid];
    }
    __syncthreads();
    if (tid == 0) {
        float r[8];
        #pragma unroll
        for (int o = 0; o < 8; ++o) r[o] = reg8[o];
        float res[8];
        res[0] = r[0]; res[1] = r[1]; res[2] = r[2];
        res[3] = 1.f / (1.f + __expf(-r[3])) - 0.5f;
        res[4] = 1.f / (1.f + __expf(-r[4])) - 0.5f;
        res[5] = 1.f / (1.f + __expf(-r[5])) - 0.5f;
        float nrm = fmaxf(sqrtf(r[6] * r[6] + r[7] * r[7]), 1e-12f);
        res[6] = r[6] / nrm; res[7] = r[7] / nrm;
        for (int o = 0; o < 8; ++o) out[b * 400 + o * 50 + k] = res[o];
    }
}

// ---------------------------------------------------------------------------
extern "C" void kernel_launch(void* const* d_in, const int* in_sizes, int n_in,
                              void* d_out, int out_size, void* d_ws, size_t ws_size,
                              hipStream_t stream) {
    const float* up16   = (const float*)d_in[0];
    const float* up8    = (const float*)d_in[1];
    const float* up4    = (const float*)d_in[2];
    const float* cls_w1 = (const float*)d_in[3];
    const float* cls_b1 = (const float*)d_in[4];
    const float* bn1_g  = (const float*)d_in[5];
    const float* bn1_b  = (const float*)d_in[6];
    const float* bn1_m  = (const float*)d_in[7];
    const float* bn1_v  = (const float*)d_in[8];
    const float* cls_w2 = (const float*)d_in[9];
    const float* cls_b2 = (const float*)d_in[10];
    const float* reg_w1 = (const float*)d_in[11];
    const float* reg_b1 = (const float*)d_in[12];
    const float* bn2_g  = (const float*)d_in[13];
    const float* bn2_b  = (const float*)d_in[14];
    const float* bn2_m  = (const float*)d_in[15];
    const float* bn2_v  = (const float*)d_in[16];
    const float* box_w  = (const float*)d_in[17];
    const float* box_b  = (const float*)d_in[18];
    float* out = (float*)d_out;

    // Workspace carve-up (256B aligned).
    char* ws = (char*)d_ws;
    size_t off = 0;
    auto carve = [&](size_t bytes) {
        size_t o = off;
        off = (off + bytes + 255) & ~(size_t)255;
        return o;
    };
    _Float16* Wpack   = (_Float16*)(ws + carve(32 * 18 * 512 * sizeof(_Float16)));
    float*    hclass  = (float*)(ws + carve(8 * 3 * 96 * 320 * sizeof(float)));
    float*    nmsbuf  = (float*)(ws + carve(8 * 3 * 96 * 320 * sizeof(float)));
    float*    tsc     = (float*)(ws + carve(8 * 3 * 50 * sizeof(float)));
    int*      tix     = (int*)(ws + carve(8 * 3 * 50 * sizeof(int)));
    int*      slot_y  = (int*)(ws + carve(1600 * sizeof(int)));
    int*      slot_x  = (int*)(ws + carve(1600 * sizeof(int)));
    float*    lxv     = (float*)(ws + carve(400 * sizeof(float)));
    float*    lyv     = (float*)(ws + carve(400 * sizeof(float)));
    float*    cf      = (float*)(ws + carve(1600 * 256 * sizeof(float)));

    pack_weights<<<(32 * 18 * 512) / 256, 256, 0, stream>>>(cls_w1, reg_w1, Wpack);

    conv_cls_fused<<<dim3(10, 96, 8), 256, 0, stream>>>(
        up4, Wpack, cls_b1, bn1_g, bn1_b, bn1_m, bn1_v, cls_w2, cls_b2, hclass);

    nms_kernel<<<(8 * 3 * 96 * 320) / 256, 256, 0, stream>>>(hclass, nmsbuf);

    topk_per_class<<<24, 256, 0, stream>>>(nmsbuf, tsc, tix);

    combine_topk<<<8, 32, 0, stream>>>(tsc, tix, out);

    build_corners<<<2, 256, 0, stream>>>(out, slot_y, slot_x, lxv, lyv);

    conv_reg_corners<<<dim3(100, 2), 256, 0, stream>>>(
        up4, Wpack, slot_y, slot_x, reg_b1, bn2_g, bn2_b, bn2_m, bn2_v, cf);

    final_head<<<400, 256, 0, stream>>>(cf, up8, up16, lxv, lyv, box_w, box_b, out);
}
